// Mamba_30889404793195
// MI455X (gfx1250) — compile-verified, matrix operand order checked by reading, not run on hardware
//
#include <hip/hip_runtime.h>
#include <hip/hip_bf16.h>
#include <math.h>

// ---------------------------------------------------------------------------
// Problem dims (fixed by the reference)
// ---------------------------------------------------------------------------
#define DM    768      // D_MODEL
#define DI    1536     // D_INNER
#define DST   16       // D_STATE
#define DTR   48       // DT_RANK
#define DCONV 4
#define BB    2
#define LL    2048
#define MTOT  (BB*LL)  // 4096 rows for all GEMMs

typedef __bf16 bf16;
typedef __attribute__((ext_vector_type(8)))  __bf16 bf16x8;
typedef __attribute__((ext_vector_type(16))) __bf16 v16bf;
typedef __attribute__((ext_vector_type(8)))  float  v8f;
typedef __attribute__((ext_vector_type(4)))  int    i32x4;

// ---------------------------------------------------------------------------
// CDNA5 async global->LDS copy (ASYNCcnt-tracked), with graceful fallback
// ROCm 7.2 signature: (v4i AS(1)* src, v4i AS(3)* lds, Ii offset, Ii cpol)
// ---------------------------------------------------------------------------
#if __has_builtin(__builtin_amdgcn_global_load_async_to_lds_b128)
#define USE_ASYNC_LDS 1
#else
#define USE_ASYNC_LDS 0
#endif

__device__ __forceinline__ void cp16_g2l(bf16* lds_dst, const bf16* gsrc) {
#if USE_ASYNC_LDS
  __builtin_amdgcn_global_load_async_to_lds_b128(
      (__attribute__((address_space(1))) i32x4*)gsrc,
      (__attribute__((address_space(3))) i32x4*)lds_dst, 0, 0);
#else
  *reinterpret_cast<bf16x8*>(lds_dst) = *reinterpret_cast<const bf16x8*>(gsrc);
#endif
}

__device__ __forceinline__ void async_join() {
#if USE_ASYNC_LDS
#if __has_builtin(__builtin_amdgcn_s_wait_asynccnt)
  __builtin_amdgcn_s_wait_asynccnt(0);
#else
  asm volatile("s_wait_asynccnt 0" ::: "memory");
#endif
#endif
}

__device__ __forceinline__ bf16 f2bf(float f) {
  unsigned u = __builtin_bit_cast(unsigned, f);
  u += 0x7fffu + ((u >> 16) & 1u);                 // round-to-nearest-even
  unsigned short h = (unsigned short)(u >> 16);
  return __builtin_bit_cast(bf16, h);
}

__device__ __forceinline__ float siluf(float x) {
  return x / (1.f + __expf(-x));
}

// ---------------------------------------------------------------------------
// Elementwise fp32 -> bf16 convert
// ---------------------------------------------------------------------------
__global__ void k_cvt_bf16(const float* __restrict__ s, bf16* __restrict__ d, int n) {
  int i = blockIdx.x * blockDim.x + threadIdx.x;
  if (i < n) d[i] = f2bf(s[i]);
}

// Extract dt-rank columns (first 48 of 80) of x_dbl as bf16, lda=48
__global__ void k_extract_dt(const float* __restrict__ xdbl, bf16* __restrict__ dt, int n) {
  int i = blockIdx.x * blockDim.x + threadIdx.x;
  if (i < n) {
    int r = i / DTR, c = i - r * DTR;
    dt[i] = f2bf(xdbl[r * (DTR + 2 * DST) + c]);
  }
}

// ---------------------------------------------------------------------------
// WMMA bf16 GEMM:  C[M,N] = A[M,Kb] * B[Kb,N]   (row-major, fp32 accum)
//   block = 256 threads = 8 waves; wave w owns rows [w*16, w*16+16)
//   tile: BM=128, BK=32, BN=NT*16; double-buffered LDS, async global->LDS
//   staging for A pipelined one K-tile ahead of the WMMAs.
//   KG: enable K-bound guards (only the K=48 delta GEMM needs them;
//       callers with KG=false guarantee Kpad == Kb).
//   EP: 0 = plain fp32 store, 1 = +bias then softplus (delta path)
// ---------------------------------------------------------------------------
template <int NT, int EP, bool KG>
__global__ __launch_bounds__(256) void k_gemm_bf16(
    const bf16* __restrict__ A, const bf16* __restrict__ B,
    float* __restrict__ C, const float* __restrict__ bias,
    int Kpad, int Kb, int lda, int ldb, int ldc)
{
  constexpr int BM = 128, BK = 32, BN = NT * 16, LDP = BK + 8; // +16B row pad
  __shared__ bf16 As[2][BM][LDP];
  __shared__ bf16 Bs[2][BN][LDP];   // stored transposed: Bs[buf][n][k]

  const int tid  = threadIdx.x;
  const int lane = tid & 31;
  const int wave = tid >> 5;
  const int tM   = blockIdx.x * BM;
  const int tN   = blockIdx.y * BN;

  // ISA 16-bit fragment layout: lanes 0-15 hold K {0-7,16-23} of row (lane),
  // lanes 16-31 hold K {8-15,24-31} of row (lane-16).  Same pattern for B cols.
  const int half = lane >> 4;          // 0 or 1
  const int kb   = half * 8;           // fragment K sub-offset
  const int mrow = wave * 16 + (lane & 15);

  // Stage one BMxBK / BKxBN tile pair into LDS buffer `buf`.
  auto stage = [&](int k0, int buf) {
    // A: 16B chunks, coalesced; async direct-to-LDS
    #pragma unroll
    for (int c = tid; c < BM * 4; c += 256) {
      const int r  = c >> 2;
      const int kc = (c & 3) * 8;
      bf16* dst = &As[buf][r][kc];
      if (!KG || (k0 + kc < Kb)) {
        cp16_g2l(dst, A + (size_t)(tM + r) * lda + k0 + kc);
      } else {
        bf16x8 z = {};
        *reinterpret_cast<bf16x8*>(dst) = z;
      }
    }
    // B: one coalesced 16B global load per 8 columns, transpose via LDS stores
    #pragma unroll
    for (int c = tid; c < BK * (BN / 8); c += 256) {
      const int kk = c / (BN / 8);
      const int nb = (c - kk * (BN / 8)) * 8;
      bf16x8 v = {};
      if (!KG || (k0 + kk < Kb))
        v = *reinterpret_cast<const bf16x8*>(B + (size_t)(k0 + kk) * ldb + tN + nb);
      #pragma unroll
      for (int j = 0; j < 8; ++j) Bs[buf][nb + j][kk] = v[j];
    }
  };

  v8f acc[NT] = {};
  const int nk = Kpad / BK;

  stage(0, 0);
  for (int i = 0; i < nk; ++i) {
    async_join();           // staged tile (issued last iteration) is in LDS
    __syncthreads();        // ... for every wave
    if (i + 1 < nk) stage((i + 1) * BK, (i + 1) & 1);  // prefetch next tile

    const int buf = i & 1;
    bf16x8 alo = *reinterpret_cast<const bf16x8*>(&As[buf][mrow][kb]);
    bf16x8 ahi = *reinterpret_cast<const bf16x8*>(&As[buf][mrow][kb + 16]);
    v16bf a = __builtin_shufflevector(alo, ahi,
        0, 1, 2, 3, 4, 5, 6, 7, 8, 9, 10, 11, 12, 13, 14, 15);

    #pragma unroll
    for (int nt = 0; nt < NT; ++nt) {
      const int nc = nt * 16 + (lane & 15);
      bf16x8 blo = *reinterpret_cast<const bf16x8*>(&Bs[buf][nc][kb]);
      bf16x8 bhi = *reinterpret_cast<const bf16x8*>(&Bs[buf][nc][kb + 16]);
      v16bf b = __builtin_shufflevector(blo, bhi,
          0, 1, 2, 3, 4, 5, 6, 7, 8, 9, 10, 11, 12, 13, 14, 15);
      acc[nt] = __builtin_amdgcn_wmma_f32_16x16x32_bf16(
          /*neg_a=*/false, a, /*neg_b=*/false, b,
          /*c_mod=*/(short)0, acc[nt], /*reuse_a=*/false, /*reuse_b=*/false);
    }
  }

  // ---- epilogue: D layout -> (row = r + 8*half, col = lane&15) per VGPR r ----
  #pragma unroll
  for (int nt = 0; nt < NT; ++nt) {
    #pragma unroll
    for (int r = 0; r < 8; ++r) {
      const int row = tM + wave * 16 + r + half * 8;
      const int col = tN + nt * 16 + (lane & 15);
      float v = acc[nt][r];
      if (EP == 1) {                       // bias + softplus (delta path)
        v += bias[col];
        v = (v > 20.f) ? v : __logf(1.f + __expf(v));
      }
      C[(size_t)row * ldc + col] = v;
    }
  }
}

// ---------------------------------------------------------------------------
// Depthwise causal conv(4) + bias + silu on the u half of xr = [u | res]
// Emits u (fp32, for the scan) and ub (bf16, feeds the W_x GEMM).
// ---------------------------------------------------------------------------
__global__ void k_conv_silu(const float* __restrict__ xr,
                            const float* __restrict__ cw,
                            const float* __restrict__ cb,
                            float* __restrict__ u, bf16* __restrict__ ub)
{
  const int i = blockIdx.x * blockDim.x + threadIdx.x;
  if (i >= MTOT * DI) return;
  const int d = i % DI;
  const int m = i / DI;
  const int l = m & (LL - 1);
  const int b = m >> 11;                        // LL = 2048 = 2^11
  float acc = cb[d];
  #pragma unroll
  for (int j = 0; j < DCONV; ++j) {
    const int ls = l - (DCONV - 1) + j;
    if (ls >= 0)
      acc += cw[d * DCONV + j] * xr[((size_t)(b * LL + ls)) * (2 * DI) + d];
  }
  const float s = siluf(acc);
  u[i]  = s;
  ub[i] = f2bf(s);
}

// ---------------------------------------------------------------------------
// Selective scan.  One 16-lane half-wave per (batch, channel): lane n holds
// state s[n]; per step: s = exp(delta*A[n])*s + delta*B[n]*u, y = sum_n s*C[n]
// via shfl_xor butterfly; fused +u*Dp and *silu(res); emits bf16 y for W_out.
// ---------------------------------------------------------------------------
__global__ __launch_bounds__(128) void k_scan(
    const float* __restrict__ delta, const float* __restrict__ u,
    const float* __restrict__ xdbl,  const float* __restrict__ xr,
    const float* __restrict__ A_log, const float* __restrict__ Dp,
    bf16* __restrict__ yb)
{
  const int tid = threadIdx.x;
  const int n   = tid & 15;
  const int ch  = blockIdx.x * 8 + (tid >> 4);    // 0 .. 3071
  const int b   = ch / DI;
  const int d   = ch - b * DI;

  const float An  = -__expf(A_log[d * DST + n]);  // A = -exp(A_log)
  const float dpd = Dp[d];
  const size_t base = (size_t)b * LL;

  float s = 0.f;
  for (int t = 0; t < LL; ++t) {
    const size_t m  = base + t;
    const float dv  = delta[m * DI + d];          // broadcast across 16 lanes
    const float uv  = u[m * DI + d];
    const float Bn  = xdbl[m * (DTR + 2 * DST) + DTR + n];
    const float Cn  = xdbl[m * (DTR + 2 * DST) + DTR + DST + n];
    s = __expf(dv * An) * s + dv * Bn * uv;
    float p = s * Cn;
    p += __shfl_xor(p, 1, 16);
    p += __shfl_xor(p, 2, 16);
    p += __shfl_xor(p, 4, 16);
    p += __shfl_xor(p, 8, 16);
    if (n == 0) {
      float y = p + uv * dpd;
      const float r = xr[m * (2 * DI) + DI + d];  // res half
      y *= siluf(r);
      yb[m * DI + d] = f2bf(y);
    }
  }
}

// ---------------------------------------------------------------------------
// Host-side pipeline
// ---------------------------------------------------------------------------
extern "C" void kernel_launch(void* const* d_in, const int* in_sizes, int n_in,
                              void* d_out, int out_size, void* d_ws, size_t ws_size,
                              hipStream_t stream)
{
  (void)in_sizes; (void)n_in; (void)out_size; (void)ws_size;
  const float* x      = (const float*)d_in[0];
  const float* W_in   = (const float*)d_in[1];
  const float* conv_w = (const float*)d_in[2];
  const float* conv_b = (const float*)d_in[3];
  const float* W_x    = (const float*)d_in[4];
  const float* W_dt   = (const float*)d_in[5];
  const float* b_dt   = (const float*)d_in[6];
  const float* A_log  = (const float*)d_in[7];
  const float* Dp     = (const float*)d_in[8];
  const float* W_out  = (const float*)d_in[9];
  float* out = (float*)d_out;

  // bump allocator over workspace (256B aligned)
  char* p = (char*)d_ws;
  auto alloc = [&](size_t bytes) -> void* {
    void* r = (void*)p; p += (bytes + 255) & ~(size_t)255; return r;
  };
  bf16*  xb     = (bf16*) alloc((size_t)MTOT * DM * 2);
  bf16*  W_inb  = (bf16*) alloc((size_t)DM * 2 * DI * 2);
  bf16*  W_xb   = (bf16*) alloc((size_t)DI * (DTR + 2 * DST) * 2);
  bf16*  W_dtb  = (bf16*) alloc((size_t)DTR * DI * 2);
  bf16*  W_outb = (bf16*) alloc((size_t)DI * DM * 2);
  float* xr     = (float*)alloc((size_t)MTOT * 2 * DI * 4);       // [u | res]
  float* u      = (float*)alloc((size_t)MTOT * DI * 4);
  bf16*  ub     = (bf16*) alloc((size_t)MTOT * DI * 2);
  float* xdbl   = (float*)alloc((size_t)MTOT * (DTR + 2 * DST) * 4);
  bf16*  dt_in  = (bf16*) alloc((size_t)MTOT * DTR * 2);
  float* delta  = (float*)alloc((size_t)MTOT * DI * 4);
  bf16*  yb     = (bf16*) alloc((size_t)MTOT * DI * 2);

  auto cdiv = [](long a, long b) { return (int)((a + b - 1) / b); };

  // 0) bf16 copies of activations / weights
  k_cvt_bf16<<<cdiv((long)MTOT * DM, 256), 256, 0, stream>>>(x, xb, MTOT * DM);
  k_cvt_bf16<<<cdiv((long)DM * 2 * DI, 256), 256, 0, stream>>>(W_in, W_inb, DM * 2 * DI);
  k_cvt_bf16<<<cdiv((long)DI * (DTR + 2 * DST), 256), 256, 0, stream>>>(W_x, W_xb, DI * (DTR + 2 * DST));
  k_cvt_bf16<<<cdiv((long)DTR * DI, 256), 256, 0, stream>>>(W_dt, W_dtb, DTR * DI);
  k_cvt_bf16<<<cdiv((long)DI * DM, 256), 256, 0, stream>>>(W_out, W_outb, DI * DM);

  // 1) xr = x @ W_in            (4096 x 3072, K=768)
  k_gemm_bf16<4, 0, false><<<dim3(MTOT / 128, (2 * DI) / 64), 256, 0, stream>>>(
      xb, W_inb, xr, nullptr, DM, DM, DM, 2 * DI, 2 * DI);

  // 2) u = silu(conv(u_half) + b)
  k_conv_silu<<<cdiv((long)MTOT * DI, 256), 256, 0, stream>>>(xr, conv_w, conv_b, u, ub);

  // 3) x_dbl = u @ W_x          (4096 x 80, K=1536)
  k_gemm_bf16<5, 0, false><<<dim3(MTOT / 128, 1), 256, 0, stream>>>(
      ub, W_xb, xdbl, nullptr, DI, DI, DI, DTR + 2 * DST, DTR + 2 * DST);

  // 4) delta = softplus(x_dbl[:, :48] @ W_dt + b_dt)   (K=48 padded to 64)
  k_extract_dt<<<cdiv((long)MTOT * DTR, 256), 256, 0, stream>>>(xdbl, dt_in, MTOT * DTR);
  k_gemm_bf16<4, 1, true><<<dim3(MTOT / 128, DI / 64), 256, 0, stream>>>(
      dt_in, W_dtb, delta, b_dt, 64, DTR, DTR, DI, DI);

  // 5) selective scan (fused +u*Dp and *silu(res)), bf16 output for W_out
  k_scan<<<(BB * DI) / 8, 128, 0, stream>>>(delta, u, xdbl, xr, A_log, Dp, yb);

  // 6) out = y @ W_out          (4096 x 768, K=1536)
  k_gemm_bf16<4, 0, false><<<dim3(MTOT / 128, DM / 64), 256, 0, stream>>>(
      yb, W_outb, out, nullptr, DI, DI, DI, DM, DM);
}
